// CorefMergeLayer_45784351375659
// MI455X (gfx1250) — matrix operand order repaired and checked by reading, not compiled, not run on hardware
//
#include <hip/hip_runtime.h>

// ---------------- problem constants ----------------
#define S_ 1024
#define B_ 16
#define H_ 1024
#define K_ 8
#define C_ 16
#define MROWS 2048            // B*K*C

typedef __attribute__((ext_vector_type(16))) __bf16 v16bf;
typedef __attribute__((ext_vector_type(8)))  float  v8f;

union Frag16 { uint4 u[2]; v16bf v; };

static __device__ __forceinline__ unsigned short f32_to_bf16(float f) {
    union { float f; unsigned u; } x; x.f = f;
    unsigned u = x.u;
    u += 0x7FFFu + ((u >> 16) & 1u);      // round-to-nearest-even
    return (unsigned short)(u >> 16);
}

// ---------------- f32 [Kd][N] -> bf16 transposed [N][Kd] ----------------
__global__ __launch_bounds__(256)
void cvt_t_bf16_kernel(const float* __restrict__ src, unsigned short* __restrict__ dst,
                       int Kd, int N) {
    __shared__ float tile[32][33];
    const int n0 = blockIdx.x * 32, k0 = blockIdx.y * 32;
    const int tx = threadIdx.x & 31, ty = threadIdx.x >> 5;   // ty 0..7
    #pragma unroll
    for (int i = 0; i < 4; ++i) {
        int r = ty + i * 8;
        tile[r][tx] = src[(size_t)(k0 + r) * N + n0 + tx];
    }
    __syncthreads();
    #pragma unroll
    for (int i = 0; i < 4; ++i) {
        int r = ty + i * 8;
        dst[(size_t)(n0 + r) * Kd + k0 + tx] = f32_to_bf16(tile[tx][r]);
    }
}

// ---------------- gather clusters: m_bank[S,B,H] -> cl[2048,H] (f32 + bf16) ----------------
__global__ __launch_bounds__(256)
void gather_kernel(const float* __restrict__ m_bank, const int* __restrict__ posi,
                   float* __restrict__ cl_f32, unsigned short* __restrict__ cl_bf16) {
    int rr = blockIdx.x;                 // (b*K + k)*C + c
    int b  = rr >> 7;                    // / (K*C)
    int pos = posi[rr];
    const float4* src = (const float4*)(m_bank + ((size_t)pos * B_ + b) * H_);
    float4 v = src[threadIdx.x];
    ((float4*)(cl_f32 + (size_t)rr * H_))[threadIdx.x] = v;
    ushort4 o;
    o.x = f32_to_bf16(v.x); o.y = f32_to_bf16(v.y);
    o.z = f32_to_bf16(v.z); o.w = f32_to_bf16(v.w);
    ((ushort4*)(cl_bf16 + (size_t)rr * H_))[threadIdx.x] = o;
}

// ---------------- scatter merged rows into output copy of m_bank ----------------
__global__ __launch_bounds__(256)
void scatter_kernel(const float* __restrict__ mrg, const int* __restrict__ posi,
                    float* __restrict__ out) {
    int rr = blockIdx.x;
    int b  = rr >> 7;
    int pos = posi[rr];
    float4 v = ((const float4*)(mrg + (size_t)rr * H_))[threadIdx.x];
    ((float4*)(out + ((size_t)pos * B_ + b) * H_))[threadIdx.x] = v;
}

// ---------------- generic bf16 WMMA GEMM ----------------
// D[M,N] = act( concat_k(A1,A2)[M,Kd] * W[Kd,N] + bias ), W given pre-transposed
// as Wt[N][Kd] bf16. A1/A2 row stride = lda (bf16), split at k == K1.
// Block tile 128x128, 8 waves, wave tile 32x64 (2x4 wmma 16x16x32 bf16).
// Double-buffered LDS; staging is pure b128 traffic (no scalar DS stores).
#define BK   32
#define ASTR 40   // LDS row stride (halves) with padding
__global__ __launch_bounds__(256)
void gemm_bf16_kernel(const unsigned short* __restrict__ A1,
                      const unsigned short* __restrict__ A2, int K1,
                      const unsigned short* __restrict__ Wt,   // [N][Kd] bf16
                      const float* __restrict__ bias,
                      float* __restrict__ outF, unsigned short* __restrict__ outB,
                      int M, int N, int Kd, int lda, int doTanh) {
    __shared__ unsigned short As[2][128 * ASTR];   // 128 x 32 A tiles
    __shared__ unsigned short Bs[2][128 * ASTR];   // 128 x 32 Wt tiles (n-major)

    const int t    = threadIdx.x;
    const int lane = t & 31;
    const int wv   = t >> 5;
    const int wm   = wv & 3;      // 0..3 -> 32-row slice
    const int wn   = wv >> 2;     // 0..1 -> 64-col slice
    const int m0   = blockIdx.y * 128;
    const int n0   = blockIdx.x * 128;

    v8f acc[2][4];
    #pragma unroll
    for (int a = 0; a < 2; ++a)
        #pragma unroll
        for (int b = 0; b < 4; ++b)
            acc[a][b] = (v8f){0.f,0.f,0.f,0.f,0.f,0.f,0.f,0.f};

    const int srow = t >> 1;               // 128 rows x 2 threads (A and B staging)
    const int scg  = (t & 1) * 16;         // 16-halves column group

    const int nk = Kd / BK;

    auto stage = [&](int kt, int buf) {
        const int kk0 = kt * BK;
        const unsigned short* Ap = (kk0 < K1) ? A1 : A2;
        const int kloc = (kk0 < K1) ? kk0 : (kk0 - K1);
        const uint4* asrc = (const uint4*)(Ap + (size_t)(m0 + srow) * lda + kloc + scg);
        const uint4* bsrc = (const uint4*)(Wt + (size_t)(n0 + srow) * Kd + kk0 + scg);
        __builtin_prefetch((const void*)(asrc + 4), 0, 0);   // next-next A tile
        __builtin_prefetch((const void*)(bsrc + 4), 0, 0);   // next-next B tile
        uint4 a0 = asrc[0], a1 = asrc[1];
        uint4 b0 = bsrc[0], b1 = bsrc[1];
        uint4* ad = (uint4*)(&As[buf][srow * ASTR + scg]);
        uint4* bd = (uint4*)(&Bs[buf][srow * ASTR + scg]);
        ad[0] = a0; ad[1] = a1;
        bd[0] = b0; bd[1] = b1;
    };

    stage(0, 0);
    __syncthreads();

    for (int kt = 0; kt < nk; ++kt) {
        const int cur = kt & 1;
        if (kt + 1 < nk) stage(kt + 1, cur ^ 1);

        // fragments per documented 16-bit WMMA layouts
        Frag16 af[2], bfr[4];
        const int k0a = (lane < 16) ? 0 : 8;    // A: lanes<16 -> K 0..7 & 16..23
        #pragma unroll
        for (int ti = 0; ti < 2; ++ti) {
            int ml = wm * 32 + ti * 16 + (lane & 15);
            const uint4* p = (const uint4*)(&As[cur][ml * ASTR + k0a]);
            af[ti].u[0] = p[0];                 // K k0a..k0a+7
            af[ti].u[1] = p[2];                 // K k0a+16..k0a+23
        }
        const int k0b = (lane < 16) ? 0 : 8;    // B: lanes<16 -> K 0..7 & 16..23
        #pragma unroll
        for (int tj = 0; tj < 4; ++tj) {
            int nl = wn * 64 + tj * 16 + (lane & 15);
            const uint4* p = (const uint4*)(&Bs[cur][nl * ASTR + k0b]);
            bfr[tj].u[0] = p[0];
            bfr[tj].u[1] = p[2];
        }
        #pragma unroll
        for (int ti = 0; ti < 2; ++ti)
            #pragma unroll
            for (int tj = 0; tj < 4; ++tj)
                acc[ti][tj] = __builtin_amdgcn_wmma_f32_16x16x32_bf16(
                    false, af[ti].v, false, bfr[tj].v,
                    (short)0, acc[ti][tj], false, false);
        __syncthreads();
    }

    // epilogue: bias + optional tanh, f32 and/or bf16 store (C/D layout)
    #pragma unroll
    for (int ti = 0; ti < 2; ++ti) {
        #pragma unroll
        for (int tj = 0; tj < 4; ++tj) {
            int n = n0 + wn * 64 + tj * 16 + (lane & 15);
            float bv = bias ? bias[n] : 0.f;
            #pragma unroll
            for (int r = 0; r < 8; ++r) {
                int m = m0 + wm * 32 + ti * 16 + r + ((lane >= 16) ? 8 : 0);
                float v = acc[ti][tj][r] + bv;
                if (doTanh) v = tanhf(v);
                if (outF) outF[(size_t)m * N + n] = v;
                if (outB) outB[(size_t)m * N + n] = f32_to_bf16(v);
            }
        }
    }
}

// ---------------- additive-attention pair scores + softmax + ctx ----------------
// One block per (b,k) group (128 groups). scores_ij = sum_h v[h]*tanh(q[i,h]+k'[j,h])
// (b_attn already folded into k'). Then softmax over j, ctx[i] = sum_j alpha_ij cl[j].
__global__ __launch_bounds__(256)
void pair_ctx_kernel(const float* __restrict__ q, const float* __restrict__ kk,
                     const float* __restrict__ cl, const float* __restrict__ v_attn,
                     unsigned short* __restrict__ ctx_bf16) {
    __shared__ float qs[16 * 256];
    __shared__ float ks[16 * 256];
    __shared__ float vs[H_];
    __shared__ float sc[256];
    __shared__ float al[256];

    const int g = blockIdx.x;                        // b*K + k
    const int t = threadIdx.x;
    const size_t base = (size_t)g * C_ * H_;

    for (int i = t; i < H_; i += 256) vs[i] = v_attn[i];

    const int pi = t >> 4, pj = t & 15;              // pair (i, j)
    float acc = 0.f;
    for (int hc = 0; hc < H_; hc += 256) {
        __syncthreads();
        for (int idx = t; idx < 16 * 256; idx += 256) {
            int r = idx >> 8, h = idx & 255;
            qs[idx] = q [base + (size_t)r * H_ + hc + h];
            ks[idx] = kk[base + (size_t)r * H_ + hc + h];
        }
        __syncthreads();
        const float* qp = &qs[pi * 256];
        const float* kp = &ks[pj * 256];
        const float* vp = &vs[hc];
        #pragma unroll 4
        for (int h = 0; h < 256; ++h)
            acc += vp[h] * tanhf(qp[h] + kp[h]);
    }
    sc[t] = acc;
    __syncthreads();

    if (t < 16) {                                    // softmax row t over 16 cols
        float mx = -1e30f;
        for (int j = 0; j < 16; ++j) mx = fmaxf(mx, sc[t * 16 + j]);
        float s = 0.f;
        for (int j = 0; j < 16; ++j) { float e = __expf(sc[t * 16 + j] - mx); al[t * 16 + j] = e; s += e; }
        float inv = 1.0f / s;
        for (int j = 0; j < 16; ++j) al[t * 16 + j] *= inv;
    }
    __syncthreads();

    float ar[16];
    const int ci = t >> 4;
    const int hb = (t & 15) * 16;
    #pragma unroll
    for (int j = 0; j < 16; ++j) ar[j] = al[ci * 16 + j];

    for (int hc = 0; hc < H_; hc += 256) {
        __syncthreads();
        for (int idx = t; idx < 16 * 256; idx += 256) {   // reuse qs as cl chunk
            int r = idx >> 8, h = idx & 255;
            qs[idx] = cl[base + (size_t)r * H_ + hc + h];
        }
        __syncthreads();
        #pragma unroll
        for (int hh = 0; hh < 16; ++hh) {
            int h = hb + hh;
            float s = 0.f;
            #pragma unroll
            for (int j = 0; j < 16; ++j) s += ar[j] * qs[j * 256 + h];
            ctx_bf16[base + (size_t)ci * H_ + hc + h] = f32_to_bf16(s);
        }
    }
}

// ---------------- launcher ----------------
extern "C" void kernel_launch(void* const* d_in, const int* in_sizes, int n_in,
                              void* d_out, int out_size, void* d_ws, size_t ws_size,
                              hipStream_t stream) {
    const float* m_bank = (const float*)d_in[0];
    const int*   posi   = (const int*)  d_in[1];
    const float* Wq     = (const float*)d_in[2];
    const float* Uk     = (const float*)d_in[3];
    const float* b_attn = (const float*)d_in[4];
    const float* v_attn = (const float*)d_in[5];
    const float* W_out  = (const float*)d_in[6];
    const float* b_out  = (const float*)d_in[7];
    const float* W_mrg  = (const float*)d_in[8];
    const float* b_mrg  = (const float*)d_in[9];
    float* out = (float*)d_out;
    char*  ws  = (char*)d_ws;

    const size_t MB = 1024ull * 1024ull;
    float*          cl_f32   = (float*)(ws + 0);                    // 8 MB
    float*          q_f32    = (float*)(ws + 8 * MB);               // 8 MB (reused for mrg)
    float*          k_f32    = (float*)(ws + 16 * MB);              // 8 MB (reused for attn_h bf16)
    float*          mrg_f32  = q_f32;
    unsigned short* attnh_b  = (unsigned short*)(ws + 16 * MB);     // 4 MB (after pair stage)
    unsigned short* cl_b     = (unsigned short*)(ws + 24 * MB);     // 4 MB
    unsigned short* ctx_b    = (unsigned short*)(ws + 28 * MB);     // 4 MB
    unsigned short* WqT_b    = (unsigned short*)(ws + 32 * MB);     // 2 MB  [N][Kd]
    unsigned short* UkT_b    = (unsigned short*)(ws + 34 * MB);     // 2 MB
    unsigned short* WoutT_b  = (unsigned short*)(ws + 36 * MB);     // 4 MB
    unsigned short* WmrgT_b  = (unsigned short*)(ws + 40 * MB);     // 4 MB

    // seed output with m_bank (untouched positions keep original values)
    hipMemcpyAsync(out, m_bank, (size_t)S_ * B_ * H_ * sizeof(float),
                   hipMemcpyDeviceToDevice, stream);

    // weights -> bf16, pre-transposed to [N][Kd]
    cvt_t_bf16_kernel<<<dim3(32, 32), 256, 0, stream>>>(Wq,    WqT_b,   H_,     H_);
    cvt_t_bf16_kernel<<<dim3(32, 32), 256, 0, stream>>>(Uk,    UkT_b,   H_,     H_);
    cvt_t_bf16_kernel<<<dim3(32, 64), 256, 0, stream>>>(W_out, WoutT_b, 2 * H_, H_);
    cvt_t_bf16_kernel<<<dim3(32, 64), 256, 0, stream>>>(W_mrg, WmrgT_b, 2 * H_, H_);

    // gather mention rows
    gather_kernel<<<MROWS, 256, 0, stream>>>(m_bank, posi, cl_f32, cl_b);

    dim3 gg(H_ / 128, MROWS / 128);   // (8,16)
    // q = cl*Wq ; k' = cl*Uk + b_attn
    gemm_bf16_kernel<<<gg, 256, 0, stream>>>(cl_b, nullptr, H_, WqT_b, nullptr,
                                             q_f32, nullptr, MROWS, H_, H_, H_, 0);
    gemm_bf16_kernel<<<gg, 256, 0, stream>>>(cl_b, nullptr, H_, UkT_b, b_attn,
                                             k_f32, nullptr, MROWS, H_, H_, H_, 0);

    // pair scores + softmax + ctx (bf16)
    pair_ctx_kernel<<<B_ * K_, 256, 0, stream>>>(q_f32, k_f32, cl_f32, v_attn, ctx_b);

    // attn_h = [ctx; cl] * W_out + b_out        (bf16 out, reuses k buffer)
    gemm_bf16_kernel<<<gg, 256, 0, stream>>>(ctx_b, cl_b, H_, WoutT_b, b_out,
                                             nullptr, attnh_b, MROWS, H_, 2 * H_, H_, 0);
    // mrg = tanh([cl; attn_h] * W_mrg + b_mrg)  (f32 out, reuses q buffer)
    gemm_bf16_kernel<<<gg, 256, 0, stream>>>(cl_b, attnh_b, H_, WmrgT_b, b_mrg,
                                             mrg_f32, nullptr, MROWS, H_, 2 * H_, H_, 1);

    // scatter merged rows into output
    scatter_kernel<<<MROWS, 256, 0, stream>>>(mrg_f32, posi, out);
}
